// DINAttention_52390011076690
// MI455X (gfx1250) — compile-verified, matrix operand order checked by reading, not run on hardware
//
#include <hip/hip_runtime.h>
#include <hip/hip_bf16.h>

// ---------------------------------------------------------------------------
// DIN attention for MI455X (gfx1250, wave32, WMMA).
// B=4096, S=200, D=64, H=64, MID=32.
// Strategy: bf16 WMMA (v_wmma_f32_16x16x32_bf16) for the MLP (30 GFLOP),
// single pass over keys (210 MB @ 23.3 TB/s ~ memory floor), weights
// pre-packed into WMMA B-fragment layout in d_ws.
// ---------------------------------------------------------------------------

typedef __bf16 v16bf __attribute__((ext_vector_type(16)));
typedef float  v8f   __attribute__((ext_vector_type(8)));

struct U32x8 { uint4 lo, hi; };

__device__ __forceinline__ unsigned short f2bf(float x) {
    unsigned u = __builtin_bit_cast(unsigned, x);
    unsigned r = (u + 0x7FFFu + ((u >> 16) & 1u)) >> 16;   // round-to-nearest-even
    return (unsigned short)r;
}

__device__ __forceinline__ v16bf make_frag(uint4 a, uint4 b) {
    U32x8 s; s.lo = a; s.hi = b;
    return __builtin_bit_cast(v16bf, s);
}

// Intra-wave LDS ordering: wave32 executes lanes in lockstep; we only need the
// wave's own DS ops complete + a compiler barrier (CDNA5 split counters).
__device__ __forceinline__ void wave_lds_sync() {
    __builtin_amdgcn_wave_barrier();
    asm volatile("s_wait_dscnt 0" ::: "memory");
    __builtin_amdgcn_wave_barrier();
}

// ---------------------------------------------------------------------------
// Prep kernel: pack W1 (256x64) and W2 (64x32) into bf16 WMMA B-fragment
// layout. Fragment (frag, lane) owns 16 contiguous bf16 (32 bytes):
//   n = ntile*16 + (lane&15), half = lane>>4,
//   k(e) = kc*32 + (e<8 ? half*8+e : 16 + half*8 + (e-8))
// so the main kernel fetches each fragment with two 128-bit loads.
// ---------------------------------------------------------------------------
__global__ void din_pack_weights(const float* __restrict__ W1,
                                 const float* __restrict__ W2,
                                 unsigned short* __restrict__ w1p,
                                 unsigned short* __restrict__ w2p) {
    int idx = blockIdx.x * blockDim.x + threadIdx.x;
    if (idx < 4 * 8 * 32 * 16) {              // W1: 4 ntiles x 8 kchunks
        int e    = idx & 15;
        int lane = (idx >> 4) & 31;
        int frag = idx >> 9;
        int kc = frag & 7, nt = frag >> 3;
        int hf = lane >> 4;
        int n  = nt * 16 + (lane & 15);
        int k  = kc * 32 + ((e < 8) ? (hf * 8 + e) : (16 + hf * 8 + (e - 8)));
        w1p[idx] = f2bf(W1[k * 64 + n]);
    } else if (idx < 4 * 8 * 32 * 16 + 2 * 2 * 32 * 16) {  // W2: 2 ntiles x 2 kchunks
        int i    = idx - 4 * 8 * 32 * 16;
        int e    = i & 15;
        int lane = (i >> 4) & 31;
        int frag = i >> 9;
        int kc = frag & 1, nt = frag >> 1;
        int hf = lane >> 4;
        int n  = nt * 16 + (lane & 15);
        int k  = kc * 32 + ((e < 8) ? (hf * 8 + e) : (16 + hf * 8 + (e - 8)));
        w2p[i] = f2bf(W2[k * 32 + n]);
    }
}

// ---------------------------------------------------------------------------
// Main kernel: one block (4 waves) per batch element.
// ---------------------------------------------------------------------------
__global__ __launch_bounds__(128) void din_main(
    const float* __restrict__ query,   // [B,64]
    const float* __restrict__ keys,    // [B,200,64]
    const int*   __restrict__ mask,    // [B,200]
    const unsigned short* __restrict__ w1p,
    const unsigned short* __restrict__ w2p,
    const float* __restrict__ b1,      // [64]
    const float* __restrict__ b2v,     // [32]
    const float* __restrict__ W3,      // [32]
    const float* __restrict__ b3,      // [1]
    float* __restrict__ out_pooled,    // [B,64]
    float* __restrict__ out_weight)    // [B,200]
{
    extern __shared__ char smem[];
    float*    q_lds     = (float*)smem;                         //  64 f
    float*    score_lds = q_lds + 64;                           // 208 f
    float*    red       = score_lds + 208;                      // 128 f
    float*    pool_tmp  = red + 128;                            //  64 f
    unsigned* attn_all  = (unsigned*)(pool_tmp + 64);           // 4 waves * 2048 u32
    unsigned short* h1_all = (unsigned short*)(attn_all + 4*2048); // 4 * 1024 u16
    float*    h2_all    = (float*)(h1_all + 4 * 1024);          // 4 * 512 f

    const int tid  = threadIdx.x;
    const int lane = tid & 31;
    const int wave = tid >> 5;
    const int b    = blockIdx.x;

    unsigned*       attn = attn_all + wave * 2048;   // 16 x 256 bf16 (u32-packed)
    unsigned short* h1   = h1_all   + wave * 1024;   // 16 x 64 bf16
    float*          h2   = h2_all   + wave * 512;    // 16 x 32 f32

    const int ln15 = lane & 15;   // A-row M / D-column N
    const int hf   = lane >> 4;   // half-wave selector in fragment layouts
    const int m0   = 8 * hf;      // D-fragment row offset

    if (tid < 64) q_lds[tid] = query[(size_t)b * 64 + tid];
    __syncthreads();

    const uint4* w1p4 = (const uint4*)w1p;
    const uint4* w2p4 = (const uint4*)w2p;
    const uint4* attn4 = (const uint4*)attn;
    const uint4* h14   = (const uint4*)h1;

    // 13 tiles of 16 positions (208 padded); waves work independently.
    for (int t = wave; t < 13; t += 4) {
        const int s0 = t * 16;

        // Prefetch next tile's key rows into GL2 (speculative).
        if (t + 4 < 13) {
            int pr = s0 + 64 + (lane >> 1);
            if (pr < 200)
                __builtin_prefetch(keys + ((size_t)b * 200 + pr) * 64, 0, 1);
        }

        // ---- build attn_in bf16 tile 16x256 in LDS: [q | k | q-k | q*k] ----
        {
            const int row = lane >> 1;
            const int s   = s0 + row;
            const int srd = s < 199 ? s : 199;            // clamp: no OOB reads
            const bool vld = (s < 200);
            const float* krow = keys + ((size_t)b * 200 + srd) * 64;
            unsigned* arow = attn + row * 128 + (lane & 1) * 64;
            const int fbase = (lane & 1) * 128;
            #pragma unroll
            for (int j = 0; j < 64; ++j) {
                int f = fbase + j * 2;
                int sec = f >> 6, d = f & 63;
                float q0 = q_lds[d], q1 = q_lds[d + 1];
                float k0 = vld ? krow[d]     : 0.f;
                float k1 = vld ? krow[d + 1] : 0.f;
                float v0, v1;
                if      (sec == 0) { v0 = q0;      v1 = q1;      }
                else if (sec == 1) { v0 = k0;      v1 = k1;      }
                else if (sec == 2) { v0 = q0 - k0; v1 = q1 - k1; }
                else               { v0 = q0 * k0; v1 = q1 * k1; }
                arow[j] = (unsigned)f2bf(v0) | ((unsigned)f2bf(v1) << 16);
            }
        }
        wave_lds_sync();

        // ---- layer 1: h1 = relu(attn_in @ W1 + b1), M=16 K=256 N=64 ----
        v8f acc[4];
        #pragma unroll
        for (int nt = 0; nt < 4; ++nt) {
            float bv = b1[nt * 16 + ln15];
            #pragma unroll
            for (int r = 0; r < 8; ++r) acc[nt][r] = bv;
        }
        #pragma unroll
        for (int kc = 0; kc < 8; ++kc) {
            uint4 a0 = attn4[ln15 * 32 + kc * 4 + hf];
            uint4 a1 = attn4[ln15 * 32 + kc * 4 + hf + 2];
            v16bf af = make_frag(a0, a1);
            #pragma unroll
            for (int nt = 0; nt < 4; ++nt) {
                int fi = (nt * 8 + kc) * 32 + lane;
                v16bf bf = make_frag(w1p4[fi * 2], w1p4[fi * 2 + 1]);
                acc[nt] = __builtin_amdgcn_wmma_f32_16x16x32_bf16(
                    false, af, false, bf, (short)0, acc[nt], false, false);
            }
        }
        #pragma unroll
        for (int nt = 0; nt < 4; ++nt)
            #pragma unroll
            for (int r = 0; r < 8; ++r) {
                float v = acc[nt][r]; v = v > 0.f ? v : 0.f;
                h1[(r + m0) * 64 + nt * 16 + ln15] = f2bf(v);
            }
        wave_lds_sync();

        // ---- layer 2: h2 = relu(h1 @ W2 + b2), M=16 K=64 N=32 ----
        v8f acc2[2];
        #pragma unroll
        for (int nt = 0; nt < 2; ++nt) {
            float bv = b2v[nt * 16 + ln15];
            #pragma unroll
            for (int r = 0; r < 8; ++r) acc2[nt][r] = bv;
        }
        #pragma unroll
        for (int kc = 0; kc < 2; ++kc) {
            uint4 a0 = h14[ln15 * 8 + kc * 4 + hf];
            uint4 a1 = h14[ln15 * 8 + kc * 4 + hf + 2];
            v16bf af = make_frag(a0, a1);
            #pragma unroll
            for (int nt = 0; nt < 2; ++nt) {
                int fi = (nt * 2 + kc) * 32 + lane;
                v16bf bf = make_frag(w2p4[fi * 2], w2p4[fi * 2 + 1]);
                acc2[nt] = __builtin_amdgcn_wmma_f32_16x16x32_bf16(
                    false, af, false, bf, (short)0, acc2[nt], false, false);
            }
        }
        #pragma unroll
        for (int nt = 0; nt < 2; ++nt)
            #pragma unroll
            for (int r = 0; r < 8; ++r) {
                float v = acc2[nt][r]; v = v > 0.f ? v : 0.f;
                h2[(r + m0) * 32 + nt * 16 + ln15] = v;
            }
        wave_lds_sync();

        // ---- layer 3 + mask: score = h2 @ W3 + b3, masked to -1e9 ----
        if (lane < 16) {
            int s = s0 + lane;
            float dot = b3[0];
            #pragma unroll
            for (int j = 0; j < 32; ++j) dot += h2[lane * 32 + j] * W3[j];
            bool valid = (s < 200) && (mask[(size_t)b * 200 + s] > 0);
            score_lds[s] = valid ? dot : -1.0e9f;
        }
    }
    __syncthreads();

    // ---- masked softmax over 208 padded scores ----
    float lm = -3.0e38f;
    if (tid < 208)       lm = score_lds[tid];
    if (tid + 128 < 208) lm = fmaxf(lm, score_lds[tid + 128]);
    red[tid] = lm;
    __syncthreads();
    for (int off = 64; off > 0; off >>= 1) {
        if (tid < off) red[tid] = fmaxf(red[tid], red[tid + off]);
        __syncthreads();
    }
    float mx = red[0];
    __syncthreads();

    float e0 = 0.f, e1 = 0.f, ls = 0.f;
    if (tid < 208) {
        float sc = score_lds[tid];
        e0 = sc > -5.0e8f ? __expf(sc - mx) : 0.f;
        ls += e0;
    }
    if (tid + 128 < 208) {
        float sc = score_lds[tid + 128];
        e1 = sc > -5.0e8f ? __expf(sc - mx) : 0.f;
        ls += e1;
    }
    red[tid] = ls;
    __syncthreads();
    for (int off = 64; off > 0; off >>= 1) {
        if (tid < off) red[tid] += red[tid + off];
        __syncthreads();
    }
    float inv = 1.0f / fmaxf(red[0], 1e-6f);
    __syncthreads();

    if (tid < 208)       score_lds[tid]       = e0 * inv;
    if (tid + 128 < 208) score_lds[tid + 128] = e1 * inv;
    __syncthreads();

    if (tid < 200)       out_weight[(size_t)b * 200 + tid]       = score_lds[tid];
    if (tid + 128 < 200) out_weight[(size_t)b * 200 + tid + 128] = score_lds[tid + 128];

    // ---- pooled = weight @ keys (coalesced re-read; keys are L2-resident) ----
    const int d  = tid & 63;
    const int ph = tid >> 6;
    float pacc = 0.f;
    const float* kb = keys + (size_t)b * 200 * 64;
    for (int s = ph * 100; s < ph * 100 + 100; ++s)
        pacc += score_lds[s] * kb[(size_t)s * 64 + d];
    if (ph == 1) pool_tmp[d] = pacc;
    __syncthreads();
    if (ph == 0) out_pooled[(size_t)b * 64 + d] = pacc + pool_tmp[d];
}

// ---------------------------------------------------------------------------
extern "C" void kernel_launch(void* const* d_in, const int* in_sizes, int n_in,
                              void* d_out, int out_size, void* d_ws, size_t ws_size,
                              hipStream_t stream) {
    const float* query = (const float*)d_in[0];
    const float* keys  = (const float*)d_in[1];
    const int*   mask  = (const int*)  d_in[2];
    const float* W1    = (const float*)d_in[3];
    const float* b1    = (const float*)d_in[4];
    const float* W2    = (const float*)d_in[5];
    const float* b2    = (const float*)d_in[6];
    const float* W3    = (const float*)d_in[7];
    const float* b3    = (const float*)d_in[8];

    unsigned short* w1p = (unsigned short*)d_ws;       // 16384 bf16 = 32 KB
    unsigned short* w2p = w1p + 16384;                 //  2048 bf16 =  4 KB

    float* out_pooled = (float*)d_out;                 // [4096,64]
    float* out_weight = out_pooled + 4096 * 64;        // [4096,200]

    const int pack_total = 4 * 8 * 32 * 16 + 2 * 2 * 32 * 16; // 18432
    din_pack_weights<<<(pack_total + 255) / 256, 256, 0, stream>>>(W1, W2, w1p, w2p);

    // dynamic LDS: q(256) + score(832) + red(512) + pool(256)
    //            + attn(32768) + h1(8192) + h2(8192) = 51008 bytes (< 64 KB)
    const size_t shmem = 51008;
    din_main<<<4096, 128, shmem, stream>>>(query, keys, mask, w1p, w2p,
                                           b1, b2, W3, b3,
                                           out_pooled, out_weight);
}
